// GNN_h_l_45114336477553
// MI455X (gfx1250) — matrix operand order repaired and checked by reading, not compiled, not run on hardware
//
#include <hip/hip_runtime.h>
#include <math.h>

typedef __attribute__((ext_vector_type(16))) __bf16 v16bf;
typedef __attribute__((ext_vector_type(8)))  float  v8f;

#define F_DIM 13
#define HID   128
#define MSG   64
#define NB    16384
#define BB    8
#define EE    262144

#define K1    64    // edge layer-1 K (34 padded to 64)
#define N1    256   // concat hidden: 128 (message) + 128 (gate)
#define KN1   96    // node layer-1 K (77 padded to 96)

#define WMMA_BF16(a, b, c) \
    __builtin_amdgcn_wmma_f32_16x16x32_bf16(false, (a), false, (b), (short)0, (c), false, false)

// ---- fast activations on hardware transcendentals (v_exp_f32 / v_rcp_f32) ----
// tanh(x) = 1 - 2/(e^{2x}+1); exp2 saturation gives exact +-1 tails (rcp(inf)=0).
__device__ inline float fast_tanh(float x) {
    float t = __builtin_amdgcn_exp2f(2.885390081777927f * x);   // 2*log2(e)
    return 1.f - 2.f * __builtin_amdgcn_rcpf(t + 1.f);
}
__device__ inline float fast_sigmoid(float x) {
    return __builtin_amdgcn_rcpf(1.f + __builtin_amdgcn_exp2f(-1.4426950408889634f * x));
}

// ---- WMMA fragment loaders (bf16, 16x16x32, wave32 layouts per ISA 7.12.2) ----

// A-matrix 16x32 bf16 from LDS, row-major with element stride `stride`.
// lanes 0-15: row m0+lane, K k0+0..7 (v0..3) and k0+16..23 (v4..7)
// lanes 16-31: row m0+lane-16, K k0+8..15 and k0+24..31
__device__ inline v16bf ldsA(const __bf16* base, int stride, int m0, int k0, int lane) {
    int r = m0 + (lane & 15);
    int h = (lane >> 4) & 1;
    const __bf16* p = base + r * stride + k0 + h * 8;
    v16bf a;
#pragma unroll
    for (int i = 0; i < 8; ++i) { a[i] = p[i]; a[i + 8] = p[i + 16]; }
    return a;
}

// B-matrix 32x16 bf16 from transposed weights WT stored [N][K] row-major.
// lanes 0-15: col n0+lane, K k0+0..7 and k0+16..23 ; lanes 16-31: col n0+lane-16, K k0+8..15 and k0+24..31
__device__ inline v16bf ldB(const __bf16* WT, int kstride, int n0, int k0, int lane) {
    const __bf16* p = WT + (size_t)(n0 + (lane & 15)) * kstride + k0 + ((lane >> 4) & 1) * 8;
    v16bf b;
#pragma unroll
    for (int i = 0; i < 8; ++i) { b[i] = p[i]; b[i + 8] = p[i + 16]; }
    return b;
}

// ---------------------------------------------------------------------------

__global__ void zero_kernel(float4* p, size_t n4) {
    size_t i = (size_t)blockIdx.x * blockDim.x + threadIdx.x;
    size_t stride = (size_t)gridDim.x * blockDim.x;
    for (; i < n4; i += stride) p[i] = make_float4(0.f, 0.f, 0.f, 0.f);
}

// Transpose + convert weights to bf16 layouts: Wc1T[256][64], We2T[64][128],
// Wn1T[128][96], Wn2T[16][128]
__global__ void prep_kernel(const float* __restrict__ We1, const float* __restrict__ Ww1,
                            const float* __restrict__ We2, const float* __restrict__ Wn1,
                            const float* __restrict__ Wn2,
                            __bf16* __restrict__ Wc1T, __bf16* __restrict__ We2T,
                            __bf16* __restrict__ Wn1T, __bf16* __restrict__ Wn2T) {
    const int total = 256 * 64 + 64 * 128 + 128 * 96 + 16 * 128;
    for (int idx = blockIdx.x * blockDim.x + threadIdx.x; idx < total;
         idx += blockDim.x * gridDim.x) {
        int i = idx;
        if (i < 256 * 64) {
            int n = i / 64, k = i % 64;
            float v = 0.f;
            if (k < 34) v = (n < 128) ? We1[k * 128 + n] : Ww1[k * 128 + (n - 128)];
            Wc1T[i] = (__bf16)v;
        } else if ((i -= 256 * 64) < 64 * 128) {
            int n = i / 128, k = i % 128;
            We2T[i] = (__bf16)We2[k * 64 + n];
        } else if ((i -= 64 * 128) < 128 * 96) {
            int n = i / 96, k = i % 96;
            Wn1T[i] = (__bf16)((k < 77) ? Wn1[k * 128 + n] : 0.f);
        } else {
            i -= 128 * 96;
            int n = i / 128, k = i % 128;
            Wn2T[i] = (__bf16)((n < 13) ? Wn2[k * 13 + n] : 0.f);
        }
    }
}

// Fused edge pipeline: gather -> layer1 (concat msg+gate) -> tanh ->
// layer2 message GEMM + gate dot -> sigmoid -> atomic scatter into agg.
__global__ __launch_bounds__(256) void edge_kernel(
    const float* __restrict__ z_h, const float* __restrict__ z_l,
    const int* __restrict__ ei,
    const __bf16* __restrict__ Wc1T, const float* __restrict__ be1,
    const float* __restrict__ bw1,
    const __bf16* __restrict__ We2T, const float* __restrict__ be2,
    const float* __restrict__ Ww2, const float* __restrict__ bw2,
    float* __restrict__ agg) {
    __shared__ __align__(16) __bf16 Ainp[64][K1];              // 8 KB
    __shared__ __align__(16) __bf16 H[64][N1];                 // 32 KB
    __shared__ __align__(16) unsigned char pool[64 * MSG * 4]; // 16 KB (Fst then Mbuf)
    __shared__ float wgt[64];
    __shared__ int   src_s[64];
    __shared__ int   tgt_s[64];

    float (*Fst)[26]  = reinterpret_cast<float(*)[26]>(pool);   // gather staging
    float (*Mbuf)[MSG] = reinterpret_cast<float(*)[MSG]>(pool); // layer-2 output

    const int tid  = threadIdx.x;
    const int lane = tid & 31;
    const int wave = tid >> 5;
    const int half = lane >> 4;
    const int b    = blockIdx.x >> 12;          // E/64 = 4096 blocks per batch
    const int e0   = (blockIdx.x & 4095) * 64;

    // warm L2/WGP$ with the weight panels (global_prefetch_b8)
    __builtin_prefetch((const char*)Wc1T + ((tid & 127) << 8), 0, 3);  // 32 KB
    __builtin_prefetch((const char*)We2T + ((tid & 63) << 8), 0, 3);   // 16 KB

    // ---- phase A: edge indices
    if (tid < 64) {
        int e = e0 + tid;
        src_s[tid] = ei[(size_t)b * 2 * EE + e];
        tgt_s[tid] = ei[(size_t)b * 2 * EE + EE + e];
    }
    __syncthreads();

    // ---- phase B: stage raw 26 floats per edge (all 256 threads)
    for (int idx = tid; idx < 64 * 26; idx += 256) {
        int e = idx / 26, j = idx % 26;
        Fst[e][j] = (j < 13) ? z_h[((size_t)b * NB + src_s[e]) * F_DIM + j]
                             : z_l[((size_t)b * NB + tgt_s[e]) * F_DIM + (j - 13)];
    }
    __syncthreads();

    // ---- phase C: build padded bf16 A-matrix element-wise (all 256 threads)
    for (int idx = tid; idx < 64 * K1; idx += 256) {
        int e = idx >> 6, k = idx & 63;
        float v = 0.f;
        if (k < 26) {
            v = Fst[e][k];
        } else if (k < 29) {                      // diff
            v = Fst[e][k - 26] - Fst[e][k - 13];
        } else if (k == 29) {                     // squared distance
            float dx = Fst[e][0] - Fst[e][13], dy = Fst[e][1] - Fst[e][14],
                  dz = Fst[e][2] - Fst[e][15];
            v = dx * dx + dy * dy + dz * dz;
        } else if (k < 33) {                      // cross component k-30
            int i = k - 30, i1 = (i + 1) % 3, i2 = (i + 2) % 3;
            v = Fst[e][3 + i1] * Fst[e][16 + i2] - Fst[e][3 + i2] * Fst[e][16 + i1];
        } else if (k == 33) {                     // |cross|
            float ax = Fst[e][3], ay = Fst[e][4], az = Fst[e][5];
            float bx = Fst[e][16], by = Fst[e][17], bz = Fst[e][18];
            float cx = ay * bz - az * by, cy = az * bx - ax * bz, cz = ax * by - ay * bx;
            v = sqrtf(cx * cx + cy * cy + cz * cz);
        }
        Ainp[e][k] = (__bf16)v;
    }
    __syncthreads();

    // ---- layer 1: [64 x 64] @ [64 x 256]; per wave: nt in {wave, wave+8},
    //      B-fragments + bias hoisted, mt unrolled.
#pragma unroll
    for (int nti = 0; nti < 2; ++nti) {
        int nt = wave + nti * 8;
        v16bf b0 = ldB(Wc1T, K1, nt * 16, 0, lane);
        v16bf b1 = ldB(Wc1T, K1, nt * 16, 32, lane);
        int n = nt * 16 + (lane & 15);
        float bias = (n < 128) ? be1[n] : bw1[n - 128];
#pragma unroll
        for (int mt = 0; mt < 4; ++mt) {
            v8f acc = {};
            acc = WMMA_BF16(ldsA(&Ainp[0][0], K1, mt * 16, 0, lane), b0, acc);
            acc = WMMA_BF16(ldsA(&Ainp[0][0], K1, mt * 16, 32, lane), b1, acc);
#pragma unroll
            for (int v = 0; v < 8; ++v)
                H[mt * 16 + v + half * 8][n] = (__bf16)fast_tanh(acc[v] + bias);
        }
    }
    __syncthreads();   // H complete; pool free to become Mbuf

    // ---- layer 2 message: [64 x 128] @ [128 x 64]; fixed nt per wave,
    //      B-fragments (4 k-steps) hoisted, 2 mt tiles per wave.
    {
        int nt = wave & 3;
        int mtbase = wave >> 2;
        v16bf bf[4];
#pragma unroll
        for (int ks = 0; ks < 4; ++ks) bf[ks] = ldB(We2T, HID, nt * 16, ks * 32, lane);
        int n = nt * 16 + (lane & 15);
        float bias = be2[n];
#pragma unroll
        for (int i = 0; i < 2; ++i) {
            int mt = mtbase + i * 2;
            v8f acc = {};
#pragma unroll
            for (int ks = 0; ks < 4; ++ks)
                acc = WMMA_BF16(ldsA(&H[0][0], N1, mt * 16, ks * 32, lane), bf[ks], acc);
#pragma unroll
            for (int v = 0; v < 8; ++v)
                Mbuf[mt * 16 + v + half * 8][n] = acc[v] + bias;
        }
    }

    // ---- gate: w = sigmoid(dot(H[e][128:256], Ww2) + bw2); 4 threads/edge
    {
        int e = tid >> 2, p = tid & 3;
        float s = 0.f;
#pragma unroll
        for (int k = p * 32; k < p * 32 + 32; ++k)
            s += (float)H[e][128 + k] * Ww2[k];
        s += __shfl_xor(s, 1, 32);
        s += __shfl_xor(s, 2, 32);
        if (p == 0) wgt[e] = fast_sigmoid(s + bw2[0]);
    }
    __syncthreads();

    // ---- scatter: agg[b][tgt] += w * m   (L2-resident float atomics)
    {
        int e  = tid >> 2;
        int c0 = (tid & 3) * 16;
        float w = wgt[e];
        float* dst = agg + ((size_t)b * NB + tgt_s[e]) * MSG;
#pragma unroll
        for (int i = 0; i < 16; ++i)
            atomicAdd(&dst[c0 + i], w * Mbuf[e][c0 + i]);
    }
}

// Node MLP: [z_l(13) | agg(64)] -> 128 -> tanh -> 13
__global__ __launch_bounds__(256) void node_kernel(
    const float* __restrict__ z_l, const float* __restrict__ agg,
    const __bf16* __restrict__ Wn1T, const float* __restrict__ bn1,
    const __bf16* __restrict__ Wn2T, const float* __restrict__ bn2,
    float* __restrict__ out) {
    __shared__ __align__(16) __bf16 An[64][KN1];   // 12 KB
    __shared__ __align__(16) __bf16 Hn[64][HID];   // 16 KB

    const int tid  = threadIdx.x;
    const int lane = tid & 31;
    const int wave = tid >> 5;
    const int half = lane >> 4;
    const int b    = blockIdx.x >> 8;              // N/64 = 256 blocks per batch
    const int n0   = (blockIdx.x & 255) * 64;

    // warm L2/WGP$ with Wn1T (24 KB) + Wn2T (4 KB), contiguous in ws
    __builtin_prefetch((const char*)Wn1T + ((tid & 127) << 8), 0, 3);

    for (int idx = tid; idx < 64 * KN1; idx += 256) {
        int i = idx / KN1, k = idx % KN1;
        float val = 0.f;
        if (k < 13)      val = z_l[((size_t)b * NB + n0 + i) * F_DIM + k];
        else if (k < 77) val = agg[((size_t)b * NB + n0 + i) * MSG + (k - 13)];
        An[i][k] = (__bf16)val;
    }
    __syncthreads();

    // layer 1: [64 x 96] @ [96 x 128]; fixed nt = wave, B-fragments hoisted.
    {
        int nt = wave;
        v16bf bf[3];
#pragma unroll
        for (int ks = 0; ks < 3; ++ks) bf[ks] = ldB(Wn1T, KN1, nt * 16, ks * 32, lane);
        int n = nt * 16 + (lane & 15);
        float bias = bn1[n];
#pragma unroll
        for (int mt = 0; mt < 4; ++mt) {
            v8f acc = {};
#pragma unroll
            for (int ks = 0; ks < 3; ++ks)
                acc = WMMA_BF16(ldsA(&An[0][0], KN1, mt * 16, ks * 32, lane), bf[ks], acc);
#pragma unroll
            for (int v = 0; v < 8; ++v)
                Hn[mt * 16 + v + half * 8][n] = (__bf16)fast_tanh(acc[v] + bias);
        }
    }
    __syncthreads();

    // layer 2: [64 x 128] @ [128 x 16], 4 tiles on waves 0..3
    if (wave < 4) {
        int mt = wave;
        v8f acc = {};
#pragma unroll
        for (int ks = 0; ks < 4; ++ks) {
            v16bf a  = ldsA(&Hn[0][0], HID, mt * 16, ks * 32, lane);
            v16bf bm = ldB(Wn2T, HID, 0, ks * 32, lane);
            acc = WMMA_BF16(a, bm, acc);
        }
        int n = lane & 15;
        if (n < 13) {
            float bias = bn2[n];
#pragma unroll
            for (int v = 0; v < 8; ++v) {
                int m = mt * 16 + v + half * 8;
                out[((size_t)b * NB + n0 + m) * F_DIM + n] = acc[v] + bias;
            }
        }
    }
}

extern "C" void kernel_launch(void* const* d_in, const int* in_sizes, int n_in,
                              void* d_out, int out_size, void* d_ws, size_t ws_size,
                              hipStream_t stream) {
    const float* z_h = (const float*)d_in[0];
    const float* z_l = (const float*)d_in[1];
    const int*   ei  = (const int*)d_in[2];
    const float* We1 = (const float*)d_in[3];
    const float* be1 = (const float*)d_in[4];
    const float* We2 = (const float*)d_in[5];
    const float* be2 = (const float*)d_in[6];
    const float* Ww1 = (const float*)d_in[7];
    const float* bw1 = (const float*)d_in[8];
    const float* Ww2 = (const float*)d_in[9];
    const float* bw2 = (const float*)d_in[10];
    const float* Wn1 = (const float*)d_in[11];
    const float* bn1 = (const float*)d_in[12];
    const float* Wn2 = (const float*)d_in[13];
    const float* bn2 = (const float*)d_in[14];
    float* out = (float*)d_out;

    // workspace layout
    char*   ws   = (char*)d_ws;
    float*  agg  = (float*)ws;                                  // B*N*64 f32 = 33.5 MB
    __bf16* Wc1T = (__bf16*)(ws + (size_t)BB * NB * MSG * 4);   // [256][64]
    __bf16* We2T = Wc1T + 256 * 64;                             // [64][128]
    __bf16* Wn1T = We2T + 64 * 128;                             // [128][96]
    __bf16* Wn2T = Wn1T + 128 * 96;                             // [16][128]

    zero_kernel<<<2048, 256, 0, stream>>>((float4*)agg, (size_t)BB * NB * MSG / 4);
    prep_kernel<<<64, 256, 0, stream>>>(We1, Ww1, We2, Wn1, Wn2, Wc1T, We2T, Wn1T, Wn2T);
    edge_kernel<<<BB * (EE / 64), 256, 0, stream>>>(z_h, z_l, ei, Wc1T, be1, bw1,
                                                    We2T, be2, Ww2, bw2, agg);
    node_kernel<<<BB * (NB / 64), 256, 0, stream>>>(z_l, agg, Wn1T, bn1, Wn2T, bn2, out);
}
